// MultiHeadAttention_29308856828260
// MI455X (gfx1250) — matrix operand order repaired and checked by reading, not compiled, not run on hardware
//
#include <hip/hip_runtime.h>

// ---------------------------------------------------------------------------
// MHA forward for MI455X (gfx1250): bf16 WMMA everywhere, flash-attention core
// B=4, S=2048, D_IN=D_OUT=1024, H=16, HD=64
// ---------------------------------------------------------------------------

typedef __attribute__((ext_vector_type(16))) __bf16 v16bf;
typedef __attribute__((ext_vector_type(8)))  __bf16 v8bf;
typedef __attribute__((ext_vector_type(8)))  float  v8f;

#define BATCH 4
#define SEQ   2048
#define DIM   1024
#define NHEAD 16
#define HDIM  64
#define MTOT  (BATCH * SEQ)   // 8192 rows

__device__ __forceinline__ v8f wmma_bf16(v16bf a, v16bf b, v8f c) {
  // D = A(16x32) * B(32x16) + C(16x16 f32)
  return __builtin_amdgcn_wmma_f32_16x16x32_bf16(
      /*neg_a=*/false, a, /*neg_b=*/false, b,
      /*c_mod=*/(short)0, c, /*reuse_a=*/false, /*reuse_b=*/false);
}

__device__ __forceinline__ v16bf cat8(v8bf lo, v8bf hi) {
  return __builtin_shufflevector(lo, hi, 0, 1, 2, 3, 4, 5, 6, 7, 8, 9, 10, 11,
                                 12, 13, 14, 15);
}

// A-fragment (16x32, bf16) from row-major [rows x ld] bf16 matrix.
// lane<16: row=row0+lane,   k in {k0..k0+15}
// lane>=16: row=row0+l-16,  k in {k0+16..k0+31}
__device__ __forceinline__ v16bf load_a16x32(const __bf16* base, int ld,
                                             int row0, int k0) {
  const int l   = threadIdx.x & 31;
  const int row = row0 + (l & 15);
  const int ko  = k0 + ((l >> 4) << 4);
  const __bf16* p = base + (size_t)row * ld + ko;
  return cat8(*(const v8bf*)(p), *(const v8bf*)(p + 8));
}

// B-fragment (32x16, bf16) from the TRANSPOSE of B stored row-major:
// baseT is [N x ld] with baseT[n][k] == B[k][n].
// lane<16: col n=n0+lane, k = k0..k0+15 ; lane>=16: n=n0+l-16, k = k0+16..k0+31
__device__ __forceinline__ v16bf load_b32x16_T(const __bf16* baseT, int ld,
                                               int k0, int n0) {
  const int l  = threadIdx.x & 31;
  const int n  = n0 + (l & 15);
  const int ko = k0 + ((l >> 4) << 4);
  const __bf16* p = baseT + (size_t)n * ld + ko;
  return cat8(*(const v8bf*)(p), *(const v8bf*)(p + 8));
}

// ---------------------------------------------------------------------------
// Prep: f32 -> bf16 cast of x
// ---------------------------------------------------------------------------
__global__ void cvt_x_bf16(const float* __restrict__ x,
                           __bf16* __restrict__ xb, int n) {
  int i = blockIdx.x * blockDim.x + threadIdx.x;
  if (i < n) xb[i] = (__bf16)x[i];
}

// Prep: W[k][n] f32 -> WT[n][k] bf16 (LDS tiled transpose), 1024x1024
__global__ void wcvt_transpose(const float* __restrict__ W,
                               __bf16* __restrict__ WT) {
  __shared__ float tile[16][17];
  const int n0 = blockIdx.x * 16, k0 = blockIdx.y * 16;
  const int tx = threadIdx.x, ty = threadIdx.y;
  tile[ty][tx] = W[(size_t)(k0 + ty) * DIM + n0 + tx];
  __syncthreads();
  WT[(size_t)(n0 + ty) * DIM + k0 + tx] = (__bf16)tile[tx][ty];
}

// ---------------------------------------------------------------------------
// Fused QKV projection: y = x @ W{q,k,v}. Writes head-major Q,K and V^T.
// 32x32 tile per wave (2 A-frags x 2 B-frags -> 4 WMMA per k-step).
// grid = (MTOT/32, DIM/256, 3), block = 256 (8 waves)
// ---------------------------------------------------------------------------
__global__ void qkv_gemm(const __bf16* __restrict__ xb,
                         const __bf16* __restrict__ WT_all,
                         __bf16* __restrict__ Qh, __bf16* __restrict__ Kh,
                         __bf16* __restrict__ Vt) {
  const int wave = threadIdx.x >> 5;
  const int m0 = blockIdx.x * 32;
  const int n0 = blockIdx.y * 256 + wave * 32;
  const int z  = blockIdx.z;
  const __bf16* WT = WT_all + (size_t)z * DIM * DIM;

  v8f acc[2][2] = {};
  for (int k0 = 0; k0 < DIM; k0 += 32) {
    if (k0 + 32 < DIM) {
      __builtin_prefetch(xb + (size_t)m0 * DIM + k0 + 32, 0, 1);
      __builtin_prefetch(WT + (size_t)n0 * DIM + k0 + 32, 0, 1);
    }
    const v16bf a0 = load_a16x32(xb, DIM, m0, k0);
    const v16bf a1 = load_a16x32(xb, DIM, m0 + 16, k0);
    const v16bf b0 = load_b32x16_T(WT, DIM, k0, n0);
    const v16bf b1 = load_b32x16_T(WT, DIM, k0, n0 + 16);
    acc[0][0] = wmma_bf16(a0, b0, acc[0][0]);
    acc[0][1] = wmma_bf16(a0, b1, acc[0][1]);
    acc[1][0] = wmma_bf16(a1, b0, acc[1][0]);
    acc[1][1] = wmma_bf16(a1, b1, acc[1][1]);
  }

  const int l  = threadIdx.x & 31;
  const int rb = (l >> 4) * 8;
#pragma unroll
  for (int mi = 0; mi < 2; ++mi) {
#pragma unroll
    for (int ni = 0; ni < 2; ++ni) {
      const int col = n0 + ni * 16 + (l & 15);
      const int h = col >> 6, hd = col & 63;
#pragma unroll
      for (int r = 0; r < 8; ++r) {
        const int tok = m0 + mi * 16 + rb + r;
        const int bb = tok >> 11, s = tok & (SEQ - 1);
        const __bf16 v = (__bf16)acc[mi][ni][r];
        if (z == 0)
          Qh[((size_t)(bb * NHEAD + h) * SEQ + s) * HDIM + hd] = v;
        else if (z == 1)
          Kh[((size_t)(bb * NHEAD + h) * SEQ + s) * HDIM + hd] = v;
        else
          Vt[((size_t)(bb * NHEAD + h) * HDIM + hd) * SEQ + s] = v;
      }
    }
  }
}

// ---------------------------------------------------------------------------
// Flash attention (causal). One wave owns one 16-query tile of one (b,h).
// S^T = K*Q^T (softmax rows land per-lane), O^T = V^T * P^T.
// grid = (SEQ/16/8, BATCH*NHEAD), block = 256
// ---------------------------------------------------------------------------
__global__ void attn_fwd(const __bf16* __restrict__ Qh,
                         const __bf16* __restrict__ Kh,
                         const __bf16* __restrict__ Vt,
                         __bf16* __restrict__ ctx) {
  const int bh   = blockIdx.y;
  const int wave = threadIdx.x >> 5;
  const int q0   = (blockIdx.x * 8 + wave) * 16;
  const int l     = threadIdx.x & 31;
  const int lq    = l & 15;
  const int lhalf = l >> 4;

  const __bf16* Qb = Qh + (size_t)bh * SEQ * HDIM;
  const __bf16* Kb = Kh + (size_t)bh * SEQ * HDIM;
  const __bf16* Vb = Vt + (size_t)bh * HDIM * SEQ;

  // Q^T B-fragments (hd halves 0..31 and 32..63); reused for every key block.
  const v16bf qf0 = load_b32x16_T(Qb, HDIM, 0, q0);
  const v16bf qf1 = load_b32x16_T(Qb, HDIM, 32, q0);

  float m_run = -3.0e38f, l_run = 0.0f;
  v8f o0 = {}, o1 = {}, o2 = {}, o3 = {};
  const int qq  = q0 + lq;
  const int nkb = (q0 + 47) >> 5;  // 32-key blocks covering keys 0..q0+15

  for (int ib = 0; ib < nkb; ++ib) {
    const int kb = ib * 32;

    // ---- scores: S^T (keys x queries), two 16-key sub-tiles ----
    v8f st0 = {}, st1 = {};
    st0 = wmma_bf16(load_a16x32(Kb, HDIM, kb, 0), qf0, st0);
    st0 = wmma_bf16(load_a16x32(Kb, HDIM, kb, 32), qf1, st0);
    st1 = wmma_bf16(load_a16x32(Kb, HDIM, kb + 16, 0), qf0, st1);
    st1 = wmma_bf16(load_a16x32(Kb, HDIM, kb + 16, 32), qf1, st1);

    // ---- scale + causal mask + block row-max ----
    float s0[8], s1[8];
    float mb = -3.0e38f;
#pragma unroll
    for (int r = 0; r < 8; ++r) {
      const int key0 = kb + r + 8 * lhalf;       // sub-tile 0 key
      const int key1 = key0 + 16;                // sub-tile 1 key
      s0[r] = (key0 <= qq) ? st0[r] * 0.125f : -3.0e38f;  // 1/sqrt(64)
      s1[r] = (key1 <= qq) ? st1[r] * 0.125f : -3.0e38f;
      mb = fmaxf(mb, fmaxf(s0[r], s1[r]));
    }
    mb = fmaxf(mb, __shfl_xor(mb, 16, 32));      // combine key halves

    const float m_new = fmaxf(m_run, mb);
    const float alpha = __expf(m_run - m_new);

    float p0[8], p1[8], psum = 0.0f;
#pragma unroll
    for (int r = 0; r < 8; ++r) {
      p0[r] = __expf(s0[r] - m_new);
      p1[r] = __expf(s1[r] - m_new);
      psum += p0[r] + p1[r];
    }
    psum += __shfl_xor(psum, 16, 32);
    l_run = l_run * alpha + psum;
    m_run = m_new;

    // ---- build P^T B-fragment (32 keys x 16 queries) via lane-xor swaps ----
    v16bf pb;
#pragma unroll
    for (int r = 0; r < 8; ++r) {
      const float x0 = __shfl_xor(p0[r], 16, 32);   // partner's sub-tile-0 p
      const float x1 = __shfl_xor(p1[r], 16, 32);   // partner's sub-tile-1 p
      const float lo = lhalf ? x1 : p0[r];          // keys kb+{0..7 | 16..23}
      const float hi = lhalf ? p1[r] : x0;          // keys kb+{8..15 | 24..31}
      pb[r]     = (__bf16)lo;
      pb[8 + r] = (__bf16)hi;
    }

    // ---- rescale running output, accumulate O^T += V^T * P^T ----
#pragma unroll
    for (int r = 0; r < 8; ++r) {
      o0[r] *= alpha; o1[r] *= alpha; o2[r] *= alpha; o3[r] *= alpha;
    }
    o0 = wmma_bf16(load_a16x32(Vb, SEQ, 0,  kb), pb, o0);
    o1 = wmma_bf16(load_a16x32(Vb, SEQ, 16, kb), pb, o1);
    o2 = wmma_bf16(load_a16x32(Vb, SEQ, 32, kb), pb, o2);
    o3 = wmma_bf16(load_a16x32(Vb, SEQ, 48, kb), pb, o3);
  }

  // ---- finalize and store ctx[b, s, h*64 + hd] as bf16 ----
  const float inv = 1.0f / l_run;
  const int bb = bh >> 4, h = bh & 15;
  __bf16* crow = ctx + ((size_t)bb * SEQ + qq) * DIM + h * HDIM;
#pragma unroll
  for (int r = 0; r < 8; ++r) {
    const int rr = r + 8 * lhalf;  // hd within 16-row block
    crow[rr]      = (__bf16)(o0[r] * inv);
    crow[16 + rr] = (__bf16)(o1[r] * inv);
    crow[32 + rr] = (__bf16)(o2[r] * inv);
    crow[48 + rr] = (__bf16)(o3[r] * inv);
  }
}

// ---------------------------------------------------------------------------
// Output projection: out = ctx @ Wo + bo (f32 out), 32x32 tile per wave
// grid = (MTOT/32, DIM/256), block = 256
// ---------------------------------------------------------------------------
__global__ void out_gemm(const __bf16* __restrict__ ctx,
                         const __bf16* __restrict__ WoT,
                         const float* __restrict__ bo,
                         float* __restrict__ out) {
  const int wave = threadIdx.x >> 5;
  const int m0 = blockIdx.x * 32;
  const int n0 = blockIdx.y * 256 + wave * 32;

  v8f acc[2][2] = {};
  for (int k0 = 0; k0 < DIM; k0 += 32) {
    if (k0 + 32 < DIM) {
      __builtin_prefetch(ctx + (size_t)m0 * DIM + k0 + 32, 0, 1);
      __builtin_prefetch(WoT + (size_t)n0 * DIM + k0 + 32, 0, 1);
    }
    const v16bf a0 = load_a16x32(ctx, DIM, m0, k0);
    const v16bf a1 = load_a16x32(ctx, DIM, m0 + 16, k0);
    const v16bf b0 = load_b32x16_T(WoT, DIM, k0, n0);
    const v16bf b1 = load_b32x16_T(WoT, DIM, k0, n0 + 16);
    acc[0][0] = wmma_bf16(a0, b0, acc[0][0]);
    acc[0][1] = wmma_bf16(a0, b1, acc[0][1]);
    acc[1][0] = wmma_bf16(a1, b0, acc[1][0]);
    acc[1][1] = wmma_bf16(a1, b1, acc[1][1]);
  }

  const int l  = threadIdx.x & 31;
  const int rb = (l >> 4) * 8;
#pragma unroll
  for (int mi = 0; mi < 2; ++mi) {
#pragma unroll
    for (int ni = 0; ni < 2; ++ni) {
      const int col = n0 + ni * 16 + (l & 15);
      const float bias = bo[col];
#pragma unroll
      for (int r = 0; r < 8; ++r) {
        out[(size_t)(m0 + mi * 16 + rb + r) * DIM + col] =
            acc[mi][ni][r] + bias;
      }
    }
  }
}

// ---------------------------------------------------------------------------
// Host-side orchestration
// ---------------------------------------------------------------------------
extern "C" void kernel_launch(void* const* d_in, const int* in_sizes, int n_in,
                              void* d_out, int out_size, void* d_ws,
                              size_t ws_size, hipStream_t stream) {
  const float* x  = (const float*)d_in[0];
  const float* Wq = (const float*)d_in[1];
  const float* Wk = (const float*)d_in[2];
  const float* Wv = (const float*)d_in[3];
  const float* Wo = (const float*)d_in[4];
  const float* bo = (const float*)d_in[5];
  float* out = (float*)d_out;

  char* ws = (char*)d_ws;
  const size_t MB = (size_t)1 << 20;
  __bf16* xb    = (__bf16*)(ws);             // 16 MB : x as bf16 [8192,1024]
  __bf16* WTqkv = (__bf16*)(ws + 16 * MB);   //  6 MB : WqT|WkT|WvT bf16
  __bf16* WoT   = (__bf16*)(ws + 22 * MB);   //  2 MB : WoT bf16
  __bf16* Qh    = (__bf16*)(ws + 24 * MB);   // 16 MB : Q [b,h,s,hd]
  __bf16* Kh    = (__bf16*)(ws + 40 * MB);   // 16 MB : K [b,h,s,hd]
  __bf16* Vt    = (__bf16*)(ws + 56 * MB);   // 16 MB : V^T [b,h,hd,s]
  __bf16* ctx   = (__bf16*)(ws + 72 * MB);   // 16 MB : ctx [b,s,1024]

  // 1) casts / transposes
  const int nx = MTOT * DIM;
  cvt_x_bf16<<<nx / 256, 256, 0, stream>>>(x, xb, nx);
  dim3 tb(16, 16), tg(DIM / 16, DIM / 16);
  wcvt_transpose<<<tg, tb, 0, stream>>>(Wq, WTqkv);
  wcvt_transpose<<<tg, tb, 0, stream>>>(Wk, WTqkv + (size_t)DIM * DIM);
  wcvt_transpose<<<tg, tb, 0, stream>>>(Wv, WTqkv + (size_t)2 * DIM * DIM);
  wcvt_transpose<<<tg, tb, 0, stream>>>(Wo, WoT);

  // 2) fused QKV projection (WMMA, 32x32 wave tiles)
  qkv_gemm<<<dim3(MTOT / 32, DIM / 256, 3), 256, 0, stream>>>(xb, WTqkv, Qh,
                                                              Kh, Vt);
  // 3) causal flash attention (WMMA)
  attn_fwd<<<dim3(SEQ / 16 / 8, BATCH * NHEAD), 256, 0, stream>>>(Qh, Kh, Vt,
                                                                  ctx);
  // 4) output projection + bias (WMMA, 32x32 wave tiles)
  out_gemm<<<dim3(MTOT / 32, DIM / 256), 256, 0, stream>>>(ctx, WoT, bo, out);
}
